// VectorQuantizer_17025250361846
// MI455X (gfx1250) — matrix operand order repaired and checked by reading, not compiled, not run on hardware
//
#include <hip/hip_runtime.h>
#include <hip/hip_bf16.h>

// VQ-VAE vector quantizer for MI455X (gfx1250, wave32).
//   z_e: (32,256,32,32) f32, codebook: (1024,256) f32
//   out: z_q (8388608 f32) ++ loss (1 f32)
// Distance GEMM (32768 x 1024 x 256) via v_wmma_f32_16x16x32_f16 with a
// register double-buffered B pipeline (loads for tile nt+1 in flight while
// the 8-WMMA chain for tile nt executes).
// ws layout (needs ~17.8 MB):
//   [0, 512K)         cb16      codebook as f16, row-major (K=1024, D=256)
//   [512K, +4K)       e2[k]     sum of squares per code (f32)
//   [516K+..)         idx[n]    argmin index per pixel (i32, 128K)
//   [659456]          accum     loss accumulator (f32)
//   [1M, 1M+16M)      flat16    z_e transposed to (N=32768, D=256) f16

typedef _Float16 v8h  __attribute__((ext_vector_type(8)));
typedef _Float16 v16h __attribute__((ext_vector_type(16)));
typedef float    v8f  __attribute__((ext_vector_type(8)));

#define DIM       256
#define KCODES    1024
#define NPIX      32768
#define NELEM     8388608

#define WS_CB16   0
#define WS_E2     524288
#define WS_IDX    528384
#define WS_ACC    659456
#define WS_FLAT16 1048576

// ---- Kernel 1: codebook -> f16, e2[k] = sum(codebook[k]^2), zero accum ----
__global__ __launch_bounds__(256) void vq_prep_k(
    const float* __restrict__ cb, _Float16* __restrict__ cb16,
    float* __restrict__ e2, float* __restrict__ accum) {
  int k = blockIdx.x;          // 0..1023
  int c = threadIdx.x;         // 0..255
  float v = cb[k * DIM + c];
  cb16[k * DIM + c] = (_Float16)v;
  float p = v * v;
  #pragma unroll
  for (int off = 16; off; off >>= 1) p += __shfl_down(p, off, 32);
  __shared__ float sm[8];
  int lane = c & 31, wid = c >> 5;
  if (lane == 0) sm[wid] = p;
  __syncthreads();
  if (c == 0) {
    float s = 0.f;
    #pragma unroll
    for (int i = 0; i < 8; ++i) s += sm[i];
    e2[k] = s;
    if (k == 0) *accum = 0.f;
  }
}

// ---- Kernel 2: z_e (B,C,H,W) f32 -> flat16 (N,D) f16 via LDS tile ----
__global__ __launch_bounds__(256) void vq_transpose_k(
    const float* __restrict__ z, _Float16* __restrict__ flat16) {
  __shared__ float tile[32][33];
  int t   = blockIdx.x;          // 8192 blocks = 32(b) * 32(hw tile) * 8(d tile)
  int dt  = t & 7;
  int hwt = (t >> 3) & 31;
  int b   = t >> 8;
  int tx  = threadIdx.x;         // 0..31
  int ty  = threadIdx.y;         // 0..7
  const float* src = z + b * 262144 + (dt * 32) * 1024 + hwt * 32;
  #pragma unroll
  for (int i = 0; i < 4; ++i) {
    int d = ty + i * 8;                       // d_local
    tile[d][tx] = src[d * 1024 + tx];         // coalesced along hw
  }
  __syncthreads();
  _Float16* dst = flat16 + (size_t)(b * 1024 + hwt * 32) * DIM + dt * 32;
  #pragma unroll
  for (int i = 0; i < 4; ++i) {
    int r = ty + i * 8;                       // hw_local
    dst[(size_t)r * DIM + tx] = (_Float16)tile[tx][r];  // coalesced along d
  }
}

// ---- Kernel 3: per-16-row tile, WMMA distance GEMM + argmin over K ----
// One wave handles 16 rows of flat against all 1024 codes; B tiles are
// double-buffered in registers so WMMAs never wait on the loads they consume.
__global__ __launch_bounds__(256) void vq_argmin_k(
    const _Float16* __restrict__ flat16, const _Float16* __restrict__ cb16,
    const float* __restrict__ e2, int* __restrict__ idx) {
  int tid   = threadIdx.x;
  int lane  = tid & 31;
  int wave  = tid >> 5;
  int mbase = (blockIdx.x * 8 + wave) * 16;   // 2048 M-tiles total
  int col   = lane & 15;
  int grp   = lane >> 4;

  // A fragments: row m = mbase+col; chunk kk: half i -> K = kk*32 + (i%8) + 16*(i/8) + 8*grp
  v16h a[8];
  {
    const _Float16* ar = flat16 + (size_t)(mbase + col) * DIM;
    #pragma unroll
    for (int kk = 0; kk < 8; ++kk) {
      int d0 = kk * 32 + grp * 8;
      v8h lo = *(const v8h*)(ar + d0);
      v8h hi = *(const v8h*)(ar + d0 + 16);
      a[kk] = __builtin_shufflevector(lo, hi, 0,1,2,3,4,5,6,7,8,9,10,11,12,13,14,15);
    }
  }

  float dmin[8];
  int   imin[8];
  #pragma unroll
  for (int j = 0; j < 8; ++j) { dmin[j] = 3.4e38f; imin[j] = 0; }

  // B fragment: col = lane&15 -> codebook row nt*16+col; half i -> K = kk*32 + (i%8) + 16*(i/8) + 8*grp
  const _Float16* bbase = cb16 + (size_t)col * DIM + grp * 8;

  // Load one full 16x256 B tile (8 chunks) into a register buffer.
  auto loadB = [&](v16h* dst, int nt) {
    const _Float16* br = bbase + (size_t)nt * 16 * DIM;
    #pragma unroll
    for (int kk = 0; kk < 8; ++kk) {
      v8h lo = *(const v8h*)(br + kk * 32);
      v8h hi = *(const v8h*)(br + kk * 32 + 16);
      dst[kk] = __builtin_shufflevector(lo, hi, 0,1,2,3,4,5,6,7,8,9,10,11,12,13,14,15);
    }
  };

  // 8 chained WMMAs + argmin update for one N tile.
  auto computeTile = [&](const v16h* bb, int nt) {
    v8f acc = {0.f, 0.f, 0.f, 0.f, 0.f, 0.f, 0.f, 0.f};
    #pragma unroll
    for (int kk = 0; kk < 8; ++kk) {
      acc = __builtin_amdgcn_wmma_f32_16x16x32_f16(
          false, a[kk], false, bb[kk], (short)0, acc, false, false);
    }
    int krow = nt * 16 + col;
    float base = e2[krow];
    #pragma unroll
    for (int j = 0; j < 8; ++j) {
      float dist = base - 2.f * acc[j];        // x^2 term constant per row
      if (dist < dmin[j]) { dmin[j] = dist; imin[j] = krow; }
    }
  };

  v16h b0[8], b1[8];
  loadB(b0, 0);
  for (int nt = 0; nt < 62; nt += 2) {
    loadB(b1, nt + 1);                          // in flight during b0's WMMAs
    __builtin_prefetch(bbase + (size_t)(nt + 3) * 16 * DIM, 0, 3);
    computeTile(b0, nt);
    loadB(b0, nt + 2);                          // in flight during b1's WMMAs
    computeTile(b1, nt + 1);
  }
  loadB(b1, 63);
  computeTile(b0, 62);
  computeTile(b1, 63);

  // min-reduce across the 16 lanes of each group (xor shuffles stay in-group)
  #pragma unroll
  for (int m = 1; m < 16; m <<= 1) {
    #pragma unroll
    for (int j = 0; j < 8; ++j) {
      float od = __shfl_xor(dmin[j], m, 32);
      int   oi = __shfl_xor(imin[j], m, 32);
      if (od < dmin[j] || (od == dmin[j] && oi < imin[j])) {
        dmin[j] = od; imin[j] = oi;
      }
    }
  }
  if (col == 0) {
    #pragma unroll
    for (int j = 0; j < 8; ++j) idx[mbase + grp * 8 + j] = imin[j];  // M = j + 8*grp
  }
}

// ---- Kernel 4: gather z_q (full fp32), accumulate sum((z_q - z_e)^2) ----
// 4 elements per thread: consecutive t share c and have consecutive n.
__global__ __launch_bounds__(256) void vq_gather_k(
    const float* __restrict__ z, const float* __restrict__ cb,
    const int* __restrict__ idx, float* __restrict__ out,
    float* __restrict__ accum) {
  int t  = (blockIdx.x * 256 + threadIdx.x) * 4;  // natural (b,c,h,w) linear index
  int hw = t & 1023;
  int c  = (t >> 10) & 255;
  int b  = t >> 18;
  int n  = b * 1024 + hw;
  int4   id4 = *(const int4*)(idx + n);
  float4 z4  = *(const float4*)(z + t);
  float4 q4;
  q4.x = cb[id4.x * DIM + c];
  q4.y = cb[id4.y * DIM + c];
  q4.z = cb[id4.z * DIM + c];
  q4.w = cb[id4.w * DIM + c];
  *(float4*)(out + t) = q4;                      // z_q_st forward value == z_q
  float dx = q4.x - z4.x, dy = q4.y - z4.y, dz = q4.z - z4.z, dw = q4.w - z4.w;
  float p = dx * dx + dy * dy + dz * dz + dw * dw;
  #pragma unroll
  for (int off = 16; off; off >>= 1) p += __shfl_down(p, off, 32);
  __shared__ float sm[8];
  int lane = threadIdx.x & 31, wid = threadIdx.x >> 5;
  if (lane == 0) sm[wid] = p;
  __syncthreads();
  if (threadIdx.x == 0) {
    float s = 0.f;
    #pragma unroll
    for (int i = 0; i < 8; ++i) s += sm[i];
    atomicAdd(accum, s);                         // global_atomic_add_f32
  }
}

// ---- Kernel 5: loss = (1 + 0.25) * mse ----
__global__ void vq_finalize_k(const float* __restrict__ accum, float* __restrict__ loss) {
  *loss = 1.25f * (*accum) / (float)NELEM;
}

extern "C" void kernel_launch(void* const* d_in, const int* in_sizes, int n_in,
                              void* d_out, int out_size, void* d_ws, size_t ws_size,
                              hipStream_t stream) {
  const float* z_e = (const float*)d_in[0];
  const float* cb  = (const float*)d_in[1];
  float* out = (float*)d_out;
  char* ws = (char*)d_ws;

  _Float16* cb16   = (_Float16*)(ws + WS_CB16);
  float*    e2     = (float*)(ws + WS_E2);
  int*      idx    = (int*)(ws + WS_IDX);
  float*    accum  = (float*)(ws + WS_ACC);
  _Float16* flat16 = (_Float16*)(ws + WS_FLAT16);

  vq_prep_k<<<KCODES, 256, 0, stream>>>(cb, cb16, e2, accum);
  vq_transpose_k<<<8192, dim3(32, 8), 0, stream>>>(z_e, flat16);
  vq_argmin_k<<<NPIX / 16 / 8, 256, 0, stream>>>(flat16, cb16, e2, idx);
  vq_gather_k<<<NELEM / 1024, 256, 0, stream>>>(z_e, cb, idx, out, accum);
  vq_finalize_k<<<1, 1, 0, stream>>>(accum, out + NELEM);
}
